// _PositionAttentionModule_326417515015
// MI455X (gfx1250) — matrix operand order repaired and checked by reading, not compiled, not run on hardware
//
#include <hip/hip_runtime.h>
#include <hip/hip_bf16.h>

#define BB 8
#define CC 512
#define CO 64
#define NN 4096

typedef __attribute__((ext_vector_type(16))) __bf16 v16bf;
typedef __attribute__((ext_vector_type(8)))  float  v8f;
typedef __attribute__((ext_vector_type(4)))  unsigned u32x4;
typedef __attribute__((ext_vector_type(8)))  int      i32x8;
typedef __attribute__((ext_vector_type(4)))  int      i32x4;

union BF16x16 { v16bf v; unsigned u[8]; };

// A-matrix 16x32 bf16 fragment from LDS tile laid out [row][k] (k contiguous), row stride ld.
__device__ __forceinline__ v16bf lds_load_A16(const __bf16* base, int row0, int ld, int k0, int lane) {
  int l = lane & 15, hi = (lane >> 4) & 1;
  const __bf16* p = base + (row0 + l) * ld + k0 + hi * 16;
  BF16x16 r;
#pragma unroll
  for (int g = 0; g < 8; ++g) {
    int kb = 2 * g;
    r.u[g] = *(const unsigned*)(p + kb);
  }
  return r.v;
}

// B-matrix 32x16 bf16 fragment from LDS tile laid out [col][k] (k contiguous), row stride ld.
__device__ __forceinline__ v16bf lds_load_B16(const __bf16* base, int col0, int ld, int k0, int lane) {
  int l = lane & 15, hi = (lane >> 4) & 1;
  const __bf16* p = base + (col0 + l) * ld + k0 + hi * 16;
  BF16x16 r;
#pragma unroll
  for (int g = 0; g < 8; ++g) r.u[g] = *(const unsigned*)(p + 2 * g);
  return r.v;
}

__device__ __forceinline__ v8f wmma_bf16(v16bf a, v16bf b, v8f c) {
  return __builtin_amdgcn_wmma_f32_16x16x32_bf16(false, a, false, b, (short)0, c, false, false);
}

__device__ __forceinline__ unsigned pack_bf2(float a, float b) {
  union { __bf16 h[2]; unsigned u; } t;
  t.h[0] = (__bf16)a; t.h[1] = (__bf16)b; return t.u;
}

// TDM 2D tile load, bf16 elements, tileCols == 64 (128B rows), LDS row pitch padded to 144B.
// pad_interval code 4 = 32 dwords (128B), pad_amount code 3 = 4 dwords (16B).
__device__ __forceinline__ void tdm_load_2d(unsigned ldsAddr, const void* gaddr,
                                            int tileRows, int tileCols, long rowStrideElems) {
  unsigned long long ga = (unsigned long long)(uintptr_t)gaddr;
  u32x4 g0 = { 1u,                                  // count=1, user descriptor
               ldsAddr,                              // lds_addr (bytes)
               (unsigned)ga,
               (unsigned)(ga >> 32) | 0x80000000u }; // global_addr hi | type=2
  unsigned w0 = (1u << 16)      // data_size = 2 bytes
              | (1u << 20)      // pad_enable
              | (4u << 22)      // pad_interval = 32 dwords
              | (3u << 25);     // pad_amount   = 4 dwords
  unsigned td0 = (unsigned)tileCols, td1 = (unsigned)tileRows;
  i32x8 g1 = { (int)w0,
               (int)((td0 & 0xFFFFu) << 16),                        // tensor_dim0 lo
               (int)((td0 >> 16) | ((td1 & 0xFFFFu) << 16)),        // tensor_dim0 hi | tensor_dim1 lo
               (int)((td1 >> 16) | ((unsigned)tileCols << 16)),     // tensor_dim1 hi | tile_dim0
               (int)(unsigned)tileRows,                             // tile_dim1 (tile_dim2=0)
               (int)(unsigned)rowStrideElems,                       // tensor_dim0_stride lo32
               0, 0 };
  i32x4 z4 = { 0, 0, 0, 0 };
  i32x8 z8 = { 0, 0, 0, 0, 0, 0, 0, 0 };
  __builtin_amdgcn_tensor_load_to_lds(g0, g1, z4, z4, z8, 0);
}

// ---------------- Projection: Q and K  (out[n,o] = sum_c x[c,n] * w[o,c] + b[o]) ----------------
__global__ __launch_bounds__(256)
void pam_proj_qk(const float* __restrict__ x, const float* __restrict__ wb,
                 const float* __restrict__ bb, const float* __restrict__ wc,
                 const float* __restrict__ bc, __bf16* __restrict__ Qo, __bf16* __restrict__ Ko) {
  __shared__ __bf16 sX[64][136];   // [n][c-chunk]
  __shared__ __bf16 sWb[64][136];  // [o][c-chunk]
  __shared__ __bf16 sWc[64][136];
  int b = blockIdx.y;
  int n0 = blockIdx.x * 64;
  int tid = threadIdx.x, lane = tid & 31, w = tid >> 5;
  const float* xb = x + (size_t)b * CC * NN;
  v8f accQ[2] = {}; v8f accK[2] = {};
  int mf = w >> 1, nfb = (w & 1) * 2;
  for (int cc = 0; cc < CC; cc += 128) {
    __syncthreads();
    for (int i = tid; i < 128 * 16; i += 256) {           // x tile, float4, transposed into [n][c]
      int ci = i >> 4, n4 = (i & 15) * 4;
      float4 v = *(const float4*)&xb[(size_t)(cc + ci) * NN + n0 + n4];
      sX[n4 + 0][ci] = (__bf16)v.x; sX[n4 + 1][ci] = (__bf16)v.y;
      sX[n4 + 2][ci] = (__bf16)v.z; sX[n4 + 3][ci] = (__bf16)v.w;
    }
    for (int i = tid; i < 64 * 32; i += 256) {            // weight chunks [o][c], float4 -> bf16x4
      int oi = i >> 5, c4 = (i & 31) * 4;
      float4 vb4 = *(const float4*)&wb[oi * CC + cc + c4];
      *(uint2*)&sWb[oi][c4] = make_uint2(pack_bf2(vb4.x, vb4.y), pack_bf2(vb4.z, vb4.w));
      float4 vc4 = *(const float4*)&wc[oi * CC + cc + c4];
      *(uint2*)&sWc[oi][c4] = make_uint2(pack_bf2(vc4.x, vc4.y), pack_bf2(vc4.z, vc4.w));
    }
    __syncthreads();
#pragma unroll
    for (int ks = 0; ks < 4; ++ks) {
      v16bf a = lds_load_A16(&sX[0][0], mf * 16, 136, ks * 32, lane);
#pragma unroll
      for (int j = 0; j < 2; ++j) {
        v16bf bq = lds_load_B16(&sWb[0][0], (nfb + j) * 16, 136, ks * 32, lane);
        accQ[j] = wmma_bf16(a, bq, accQ[j]);
        v16bf bk = lds_load_B16(&sWc[0][0], (nfb + j) * 16, 136, ks * 32, lane);
        accK[j] = wmma_bf16(a, bk, accK[j]);
      }
    }
  }
  int l = lane & 15, hi = (lane >> 4) & 1;
#pragma unroll
  for (int j = 0; j < 2; ++j) {
    int o = (nfb + j) * 16 + l;
    float biasQ = bb[o], biasK = bc[o];
#pragma unroll
    for (int r = 0; r < 8; ++r) {
      int m = mf * 16 + r + hi * 8;
      size_t off = ((size_t)b * NN + n0 + m) * CO + o;
      Qo[off] = (__bf16)(accQ[j][r] + biasQ);
      Ko[off] = (__bf16)(accK[j][r] + biasK);
    }
  }
}

// ---------------- Projection: V  (out[o,n] = sum_c w_d[o,c] * x[c,n] + b_d[o]) ----------------
__global__ __launch_bounds__(256)
void pam_proj_v(const float* __restrict__ x, const float* __restrict__ wd,
                const float* __restrict__ bd, __bf16* __restrict__ Vo) {
  __shared__ __bf16 sW[64][136];   // [o][c-chunk]
  __shared__ __bf16 sX[64][136];   // [n][c-chunk]
  int b = blockIdx.z, o0 = blockIdx.y * 64, n0 = blockIdx.x * 64;
  int tid = threadIdx.x, lane = tid & 31, w = tid >> 5;
  const float* xb = x + (size_t)b * CC * NN;
  v8f acc[2] = {};
  int mf = w >> 1, nfb = (w & 1) * 2;
  for (int cc = 0; cc < CC; cc += 128) {
    __syncthreads();
    for (int i = tid; i < 64 * 32; i += 256) {
      int oi = i >> 5, c4 = (i & 31) * 4;
      float4 v4 = *(const float4*)&wd[(o0 + oi) * CC + cc + c4];
      *(uint2*)&sW[oi][c4] = make_uint2(pack_bf2(v4.x, v4.y), pack_bf2(v4.z, v4.w));
    }
    for (int i = tid; i < 128 * 16; i += 256) {
      int ci = i >> 4, n4 = (i & 15) * 4;
      float4 v = *(const float4*)&xb[(size_t)(cc + ci) * NN + n0 + n4];
      sX[n4 + 0][ci] = (__bf16)v.x; sX[n4 + 1][ci] = (__bf16)v.y;
      sX[n4 + 2][ci] = (__bf16)v.z; sX[n4 + 3][ci] = (__bf16)v.w;
    }
    __syncthreads();
#pragma unroll
    for (int ks = 0; ks < 4; ++ks) {
      v16bf a = lds_load_A16(&sW[0][0], mf * 16, 136, ks * 32, lane);
#pragma unroll
      for (int j = 0; j < 2; ++j) {
        v16bf bx = lds_load_B16(&sX[0][0], (nfb + j) * 16, 136, ks * 32, lane);
        acc[j] = wmma_bf16(a, bx, acc[j]);
      }
    }
  }
  int l = lane & 15, hi = (lane >> 4) & 1;
#pragma unroll
  for (int j = 0; j < 2; ++j) {
#pragma unroll
    for (int r = 0; r < 8; ++r) {
      int mo = o0 + mf * 16 + r + hi * 8;
      int n = n0 + (nfb + j) * 16 + l;
      Vo[((size_t)b * CC + mo) * NN + n] = (__bf16)(acc[j][r] + bd[mo]);
    }
  }
}

// ---------------- Fused flash attention + residual (TDM double-buffered tiles) ----------------
struct FlashSmem {
  __bf16 Qs[64][72];                       // [m][dim]
  __bf16 Ks[2][64][72];                    // [key][dim], double buffered
  __bf16 Vs[2][128][72];                   // [c][key],   double buffered
  union {
    struct {
      float  Ss[64][68];                   // [m][key]
      __bf16 Ps[64][72];                   // [m][key] bf16 probs
    } a;
    float Outs[64][130];                   // [m][c_local]
  } u;
  float rmax[64], rsum[64], rscale[64];
};

__global__ __launch_bounds__(256)
void pam_flash(const __bf16* __restrict__ Q, const __bf16* __restrict__ K,
               const __bf16* __restrict__ V, const float* __restrict__ x,
               const float* __restrict__ alphaPtr, float* __restrict__ out) {
  __shared__ FlashSmem sm;
  int b = blockIdx.z, c0 = blockIdx.y * 128, m0 = blockIdx.x * 64;
  int tid = threadIdx.x, lane = tid & 31, w = tid >> 5;
  int l = lane & 15, hi = (lane >> 4) & 1;

  const __bf16* Qb = Q + ((size_t)b * NN + m0) * CO;
  const __bf16* Kb = K + (size_t)b * NN * CO;
  const __bf16* Vb = V + ((size_t)b * CC + c0) * NN;

  // Q tile via TDM, then wait + publish
  if (w == 0) {
    tdm_load_2d((unsigned)(uintptr_t)&sm.Qs[0][0], Qb, 64, 64, 64);
    __builtin_amdgcn_s_wait_tensorcnt((short)0);
  }
  if (tid < 64) { sm.rmax[tid] = -1e30f; sm.rsum[tid] = 0.f; }
  __syncthreads();

  int mfS = w >> 1, nfb = (w & 1) * 2;
  v16bf aQ0 = lds_load_A16(&sm.Qs[0][0], mfS * 16, 72, 0, lane);
  v16bf aQ1 = lds_load_A16(&sm.Qs[0][0], mfS * 16, 72, 32, lane);

  // prefetch first K/V tile pair into buffer 0
  if (w == 0) {
    tdm_load_2d((unsigned)(uintptr_t)&sm.Ks[0][0][0], Kb, 64, 64, 64);
    tdm_load_2d((unsigned)(uintptr_t)&sm.Vs[0][0][0], Vb, 128, 64, NN);
  }

  v8f accO[4] = {};
  const int nTiles = NN / 64;

  for (int t = 0; t < nTiles; ++t) {
    int buf = t & 1;
    if (w == 0) {
      if (t + 1 < nTiles) {
        size_t kt2 = (size_t)(t + 1) * 64;
        tdm_load_2d((unsigned)(uintptr_t)&sm.Ks[buf ^ 1][0][0], Kb + kt2 * CO, 64, 64, 64);
        tdm_load_2d((unsigned)(uintptr_t)&sm.Vs[buf ^ 1][0][0], Vb + kt2, 128, 64, NN);
        __builtin_amdgcn_s_wait_tensorcnt((short)2);   // current pair done, next pair in flight
      } else {
        __builtin_amdgcn_s_wait_tensorcnt((short)0);
      }
    }
    __syncthreads();

    // S = Q K^T  (64x64), 16 frags split 2-per-wave
#pragma unroll
    for (int j = 0; j < 2; ++j) {
      v8f s = {};
      v16bf bk0 = lds_load_B16(&sm.Ks[buf][0][0], (nfb + j) * 16, 72, 0, lane);
      s = wmma_bf16(aQ0, bk0, s);
      v16bf bk1 = lds_load_B16(&sm.Ks[buf][0][0], (nfb + j) * 16, 72, 32, lane);
      s = wmma_bf16(aQ1, bk1, s);
#pragma unroll
      for (int r = 0; r < 8; ++r)
        sm.u.a.Ss[mfS * 16 + r + hi * 8][(nfb + j) * 16 + l] = s[r];
    }
    __syncthreads();

    // Online softmax: 4 lanes per row, 16 keys per lane
    {
      int row = tid >> 2, seg = (tid & 3) * 16;
      float vals[16];
      float mx = -1e30f;
#pragma unroll
      for (int i = 0; i < 16; ++i) { vals[i] = sm.u.a.Ss[row][seg + i]; mx = fmaxf(mx, vals[i]); }
      mx = fmaxf(mx, __shfl_xor(mx, 1));
      mx = fmaxf(mx, __shfl_xor(mx, 2));
      float om = sm.rmax[row];
      float nm = fmaxf(om, mx);
      float sum = 0.f;
#pragma unroll
      for (int i = 0; i < 16; ++i) {
        float e = __expf(vals[i] - nm);
        sum += e;
        sm.u.a.Ps[row][seg + i] = (__bf16)e;
      }
      sum += __shfl_xor(sum, 1);
      sum += __shfl_xor(sum, 2);
      if ((tid & 3) == 0) {
        float sc = __expf(om - nm);
        sm.rscale[row] = sc;
        sm.rsum[row] = sm.rsum[row] * sc + sum;
        sm.rmax[row] = nm;
      }
    }
    __syncthreads();

    // Rescale accumulators + PV: wave w owns 16 output channels [w*16, w*16+16)
    v16bf bv0 = lds_load_B16(&sm.Vs[buf][0][0], w * 16, 72, 0, lane);
    v16bf bv1 = lds_load_B16(&sm.Vs[buf][0][0], w * 16, 72, 32, lane);
#pragma unroll
    for (int mf = 0; mf < 4; ++mf) {
#pragma unroll
      for (int r = 0; r < 8; ++r)
        accO[mf][r] *= sm.rscale[mf * 16 + r + hi * 8];
      v16bf a0 = lds_load_A16(&sm.u.a.Ps[0][0], mf * 16, 72, 0, lane);
      accO[mf] = wmma_bf16(a0, bv0, accO[mf]);
      v16bf a1 = lds_load_A16(&sm.u.a.Ps[0][0], mf * 16, 72, 32, lane);
      accO[mf] = wmma_bf16(a1, bv1, accO[mf]);
    }
    __syncthreads();   // all reads of this buffer done before next iteration's TDM overwrites
  }

#pragma unroll
  for (int mf = 0; mf < 4; ++mf) {
#pragma unroll
    for (int r = 0; r < 8; ++r) {
      int m = mf * 16 + r + hi * 8;
      sm.u.Outs[m][w * 16 + l] = accO[mf][r] / sm.rsum[m];
    }
  }
  __syncthreads();

  float alpha = alphaPtr[0];
  const float* xb = x + (size_t)b * CC * NN;
  float* ob = out + (size_t)b * CC * NN;
  for (int i = tid; i < 128 * 16; i += 256) {             // float4 over m, coalesced
    int cl = i >> 4, m4 = (i & 15) * 4;
    size_t off = (size_t)(c0 + cl) * NN + m0 + m4;
    float4 xv = *(const float4*)(xb + off);
    float4 r;
    r.x = alpha * sm.u.Outs[m4 + 0][cl] + xv.x;
    r.y = alpha * sm.u.Outs[m4 + 1][cl] + xv.y;
    r.z = alpha * sm.u.Outs[m4 + 2][cl] + xv.z;
    r.w = alpha * sm.u.Outs[m4 + 3][cl] + xv.w;
    *(float4*)(ob + off) = r;
  }
}

extern "C" void kernel_launch(void* const* d_in, const int* in_sizes, int n_in,
                              void* d_out, int out_size, void* d_ws, size_t ws_size,
                              hipStream_t stream) {
  (void)in_sizes; (void)n_in; (void)out_size; (void)ws_size;
  const float* x     = (const float*)d_in[0];
  const float* wb    = (const float*)d_in[1];
  const float* bb    = (const float*)d_in[2];
  const float* wc    = (const float*)d_in[3];
  const float* bc    = (const float*)d_in[4];
  const float* wd    = (const float*)d_in[5];
  const float* bd    = (const float*)d_in[6];
  const float* alpha = (const float*)d_in[7];
  float* out = (float*)d_out;

  __bf16* Qbuf = (__bf16*)d_ws;                                   // B*N*64 bf16
  __bf16* Kbuf = Qbuf + (size_t)BB * NN * CO;                     // B*N*64 bf16
  __bf16* Vbuf = Kbuf + (size_t)BB * NN * CO;                     // B*C*N bf16

  pam_proj_qk<<<dim3(NN / 64, BB), 256, 0, stream>>>(x, wb, bb, wc, bc, Qbuf, Kbuf);
  pam_proj_v<<<dim3(NN / 64, CC / 64, BB), 256, 0, stream>>>(x, wd, bd, Vbuf);
  pam_flash<<<dim3(NN / 64, CC / 128, BB), 256, 0, stream>>>(Qbuf, Kbuf, Vbuf, x, alpha, out);
}
